// MRCLoss_89678917140921
// MI455X (gfx1250) — matrix operand order repaired and checked by reading, not compiled, not run on hardware
//
#include <hip/hip_runtime.h>

// MRC prototype-contrastive loss for MI455X (gfx1250, wave32, WMMA).
//
// Roofline: 160 MB mandatory traffic -> ~7us @ 23.3 TB/s; 8.6 GFLOP via
// f16 WMMA (524k v_wmma_f32_16x16x32_f16) is far below the matrix ceiling,
// so the kernel is HBM-bound. Row normalization is folded into the epilogue
// (dot(x/|x|,p) = dot(x,p)/|x|), norms computed from the same loads as the
// A fragments (zero extra traffic).

typedef __attribute__((ext_vector_type(16))) _Float16 v16h;
typedef __attribute__((ext_vector_type(8)))  float    v8f;

namespace {
constexpr int kN = 65536;          // samples
constexpr int kD = 512;            // feature dim (K)
constexpr int kP = 128;            // prototypes (N of GEMM)
constexpr int kTiles = kN / 16;    // 4096 row tiles of 16
constexpr int kWavesPerBlock = 8;  // 256 threads
constexpr int kBlocks = kTiles / kWavesPerBlock;  // 512
constexpr int kBHalfs = 16 * 8 * 32 * 16;         // 65536 halves = 128 KB
}

// ---------------------------------------------------------------------------
// Kernel 0: convert prototype_features (P x D f32, row major) into f16,
// pre-swizzled into the WMMA B-fragment layout (ISA 7.12.2, 16-bit B 32x16):
//   fragment index (kt, t): lane l, half h holds B(k, n) with
//     n = t*16 + (l & 15),  k = kt*32 + 8*(l>>4) + (h < 8 ? h : 8 + h)
// stored as out[((kt*8 + t)*32 + lane)*16 + h] so each lane reads its 32-byte
// fragment contiguously from LDS.
// ---------------------------------------------------------------------------
__global__ void mrc_prep_B(const float* __restrict__ proto,
                           _Float16* __restrict__ outB) {
  int o = blockIdx.x * 256 + threadIdx.x;       // 0 .. 65535
  int h    = o & 15;
  int lane = (o >> 4) & 31;
  int t    = (o >> 9) & 7;
  int kt   = o >> 12;                            // 0 .. 15
  int p = t * 16 + (lane & 15);
  int d = kt * 32 + 8 * (lane >> 4) + ((h < 8) ? h : 8 + h);
  outB[o] = (_Float16)proto[p * kD + d];
}

// ---------------------------------------------------------------------------
// Kernel 1: per-wave 16x128 logits tile via v_wmma_f32_16x16x32_f16,
// fused softmax-with-mask epilogue, deterministic block partial sums.
// ---------------------------------------------------------------------------
__global__ __launch_bounds__(256) void mrc_main(
    const float*    __restrict__ bfeat,    // (N, D) f32
    const _Float16* __restrict__ Bsw,      // pre-swizzled B fragments
    const int*      __restrict__ labels,   // (N, P) int32 {0,1}
    const float*    __restrict__ weights,  // (P,) f32
    float*          __restrict__ partials) // (kBlocks,) f32
{
  __shared__ _Float16 lb[kBHalfs];   // 128 KB of B fragments
  __shared__ float    wsum[kWavesPerBlock];

  const int tid  = threadIdx.x;
  const int wave = tid >> 5;
  const int lane = tid & 31;
  const int grp  = lane & 15;   // column group / row-of-A this lane loads
  const int hi   = lane >> 4;   // half-wave selector

  // Cooperative copy of swizzled B into LDS (16B vectors).
  {
    const uint4* src = (const uint4*)Bsw;
    uint4*       dst = (uint4*)lb;
    for (int i = tid; i < kBHalfs / 8; i += 256) dst[i] = src[i];
  }
  __syncthreads();

  // Per-lane prototype weights for its 8 columns (col = t*16 + grp).
  float wcol[8];
#pragma unroll
  for (int t = 0; t < 8; ++t) wcol[t] = weights[t * 16 + grp];

  const v16h* Bf = (const v16h*)lb;
  float tileAcc = 0.f;

  for (int tile = blockIdx.x * kWavesPerBlock + wave; tile < kTiles;
       tile += gridDim.x * kWavesPerBlock) {
    const int rowBase = tile * 16;
    const float* rowp = bfeat + (size_t)(rowBase + grp) * kD;

    v8f acc[8];
    v8f zero = {};
#pragma unroll
    for (int t = 0; t < 8; ++t) acc[t] = zero;

    float sumsq = 0.f;

    for (int kt = 0; kt < 16; ++kt) {
      // A fragment (ISA 7.12.2, 16-bit A 16x32): lane l is row grp; halves
      // 0..7 -> k = kt*32 + (hi?8:0)+0..7 ; halves 8..15 -> +16.
      const int d0 = kt * 32 + (hi ? 8 : 0);
      float4 x0 = *(const float4*)(rowp + d0);
      float4 x1 = *(const float4*)(rowp + d0 + 4);
      float4 x2 = *(const float4*)(rowp + d0 + 16);
      float4 x3 = *(const float4*)(rowp + d0 + 20);

      sumsq += x0.x * x0.x + x0.y * x0.y + x0.z * x0.z + x0.w * x0.w;
      sumsq += x1.x * x1.x + x1.y * x1.y + x1.z * x1.z + x1.w * x1.w;
      sumsq += x2.x * x2.x + x2.y * x2.y + x2.z * x2.z + x2.w * x2.w;
      sumsq += x3.x * x3.x + x3.y * x3.y + x3.z * x3.z + x3.w * x3.w;

      v16h a;
      a[0]  = (_Float16)x0.x;  a[1]  = (_Float16)x0.y;
      a[2]  = (_Float16)x0.z;  a[3]  = (_Float16)x0.w;
      a[4]  = (_Float16)x1.x;  a[5]  = (_Float16)x1.y;
      a[6]  = (_Float16)x1.z;  a[7]  = (_Float16)x1.w;
      a[8]  = (_Float16)x2.x;  a[9]  = (_Float16)x2.y;
      a[10] = (_Float16)x2.z;  a[11] = (_Float16)x2.w;
      a[12] = (_Float16)x3.x;  a[13] = (_Float16)x3.y;
      a[14] = (_Float16)x3.z;  a[15] = (_Float16)x3.w;

      const v16h* bp = Bf + (size_t)kt * (8 * 32) + lane;
#pragma unroll
      for (int t = 0; t < 8; ++t) {
        v16h b = bp[t * 32];
        acc[t] = __builtin_amdgcn_wmma_f32_16x16x32_f16(
            false, a, false, b, (short)0, acc[t], false, false);
      }
    }

    // Full-row sum of squares: lane l and l+16 together cover row `grp`.
    float ss   = sumsq + __shfl_xor(sumsq, 16, 32);
    float invn = 1.f / fmaxf(sqrtf(ss), 1e-12f);  // 1/max(|x|, eps)

    // Epilogue. C layout: vgpr v of lane l holds row (v + 8*hi), col grp of
    // each 16-col tile t. Row reductions = xor-shuffles over {1,2,4,8}.
#pragma unroll
    for (int v = 0; v < 8; ++v) {
      const int   r    = rowBase + v + 8 * hi;
      const float invr = __shfl(invn, v + 8 * hi, 32) * 10.f;  // 1/(|x|*T)
      const int* labRow = labels + (size_t)r * kP;

      float p[8], mk[8];
      float pm = -3.402823e38f;
#pragma unroll
      for (int t = 0; t < 8; ++t) {
        p[t] = acc[t][v] * invr;
        pm = fmaxf(pm, p[t]);
        mk[t] = (float)labRow[t * 16 + grp] * wcol[t];
      }
      pm = fmaxf(pm, __shfl_xor(pm, 1, 32));
      pm = fmaxf(pm, __shfl_xor(pm, 2, 32));
      pm = fmaxf(pm, __shfl_xor(pm, 4, 32));
      pm = fmaxf(pm, __shfl_xor(pm, 8, 32));

      float se = 0.f, sm = 0.f, sk = 0.f;
#pragma unroll
      for (int t = 0; t < 8; ++t) {
        float e = __expf(p[t] - pm);
        se += e * (1.f - mk[t]);       // sum exp(logits)*(1-mask)
        sm += mk[t] * (p[t] - pm);     // sum mask*logits (stabilized)
        sk += mk[t];                   // sum mask
      }
      se += __shfl_xor(se, 1, 32); se += __shfl_xor(se, 2, 32);
      se += __shfl_xor(se, 4, 32); se += __shfl_xor(se, 8, 32);
      sm += __shfl_xor(sm, 1, 32); sm += __shfl_xor(sm, 2, 32);
      sm += __shfl_xor(sm, 4, 32); sm += __shfl_xor(sm, 8, 32);
      sk += __shfl_xor(sk, 1, 32); sk += __shfl_xor(sk, 2, 32);
      sk += __shfl_xor(sk, 4, 32); sk += __shfl_xor(sk, 8, 32);

      if (grp == 0) tileAcc += sm - logf(se) * sk;  // sum_p mask*log_prob
    }
  }

  // Lanes 0 and 16 each hold half the rows' sums; fold and stage per wave.
  tileAcc += __shfl_xor(tileAcc, 16, 32);
  if (lane == 0) wsum[wave] = tileAcc;
  __syncthreads();
  if (tid == 0) {
    float s = 0.f;
    for (int w = 0; w < kWavesPerBlock; ++w) s += wsum[w];
    partials[blockIdx.x] = s;  // deterministic: fixed order
  }
}

// ---------------------------------------------------------------------------
// Kernel 2: deterministic fixed-order finalize.
// loss = lambda * (T/baseT) * (-sum mean_log_prob_pos) / N
// ---------------------------------------------------------------------------
__global__ void mrc_finalize(const float* __restrict__ partials, int n,
                             float* __restrict__ out) {
  if (blockIdx.x == 0 && threadIdx.x == 0) {
    double s = 0.0;
    for (int i = 0; i < n; ++i) s += (double)partials[i];
    out[0] = (float)(-(0.1 / (double)kN) * s);
  }
}

extern "C" void kernel_launch(void* const* d_in, const int* in_sizes, int n_in,
                              void* d_out, int out_size, void* d_ws,
                              size_t ws_size, hipStream_t stream) {
  // setup_inputs order: outputs, targets, backbone_features,
  // prototype_features, prototype_labels, prototype_weights
  const float* bfeat   = (const float*)d_in[2];
  const float* proto   = (const float*)d_in[3];
  const int*   labels  = (const int*)d_in[4];
  const float* weights = (const float*)d_in[5];

  _Float16* Bsw      = (_Float16*)d_ws;                       // 128 KB
  float*    partials = (float*)((char*)d_ws + kBHalfs * sizeof(_Float16));

  mrc_prep_B<<<kBHalfs / 256, 256, 0, stream>>>(proto, Bsw);
  mrc_main<<<kBlocks, 256, 0, stream>>>(bfeat, Bsw, labels, weights, partials);
  mrc_finalize<<<1, 1, 0, stream>>>(partials, kBlocks, (float*)d_out);
}
